// SoftAttention_52329881534730
// MI455X (gfx1250) — compile-verified, matrix operand order checked by reading, not run on hardware
//
#include <hip/hip_runtime.h>
#include <hip/hip_bf16.h>
#include <math.h>

// Sizes from the reference
#define Bz  32
#define Sz  4096
#define ENCz 512
#define DECz 512
#define ATTz 512

typedef __attribute__((ext_vector_type(16))) __bf16 v16bf;
typedef __attribute__((ext_vector_type(4)))  __bf16 v4bf;
typedef __attribute__((ext_vector_type(8)))  float  v8f;

// ---------------------------------------------------------------------------
// Kernel 1: att2[b][a] = sum_d W_hid[a][d] * dec[b][d]
// ---------------------------------------------------------------------------
__global__ __launch_bounds__(256) void att2_kernel(const float* __restrict__ dec,
                                                   const float* __restrict__ Whid,
                                                   float* __restrict__ att2) {
    int a = blockIdx.x * 256 + threadIdx.x;   // 0..511
    int b = blockIdx.y;
    const float4* wr = (const float4*)(Whid + (size_t)a * DECz);
    const float4* dr = (const float4*)(dec  + (size_t)b * DECz);
    float s = 0.f;
    for (int i = 0; i < DECz / 4; ++i) {
        float4 w = wr[i], x = dr[i];
        s = fmaf(w.x, x.x, s); s = fmaf(w.y, x.y, s);
        s = fmaf(w.z, x.z, s); s = fmaf(w.w, x.w, s);
    }
    att2[(size_t)b * ATTz + a] = s;
}

// ---------------------------------------------------------------------------
// Kernel 2 (main): scores[b][s] = W_att . relu(enc[b,s,:] @ W_enc^T + att2[b,:]) + b_att
// One wave = 16 S-rows, 8 waves/block = 128 rows. grid (S/128, B).
// A (enc rows, K=512 as bf16) lives in registers: 16 x v16bf fragments.
// Two W_enc a-tiles (2 x 16 x 512) staged in LDS per iteration -> two
// independent WMMA accumulation chains sharing the A fragments.
// ---------------------------------------------------------------------------
__global__ __launch_bounds__(256) void scores_kernel(const float* __restrict__ enc,
                                                     const float* __restrict__ Wenc,
                                                     const float* __restrict__ Watt,
                                                     const float* __restrict__ batt,
                                                     const float* __restrict__ att2,
                                                     float* __restrict__ scores) {
    // LDS: two B-fragment tiles, each [kk][lane][16 bf16] -> 2 * 16 KB
    __shared__ __align__(32) __bf16 ldsB[2][16 * 32 * 16];

    const int tid  = threadIdx.x;
    const int wave = tid >> 5;
    const int lane = tid & 31;
    const int m    = lane & 15;      // M (A/C row) and N (B/C col) index
    const int hi   = lane >> 4;      // which K-half this lane holds
    const int b    = blockIdx.y;
    const int s0   = blockIdx.x * 128 + wave * 16;

    // ---- Load A fragments (16-bit A 16x32 layout; lanes l and l+16 share row M=l)
    const float* rowp = enc + (((size_t)b * Sz) + s0 + m) * (size_t)ENCz;
    v16bf A[16];
#pragma unroll
    for (int kk = 0; kk < 16; ++kk) {
#pragma unroll
        for (int j = 0; j < 8; ++j) {
            int k32 = ((j < 4) ? (2 * j) : (16 + 2 * (j - 4))) + hi * 8;
            float2 f = *(const float2*)(rowp + kk * 32 + k32);
            A[kk][2 * j]     = (__bf16)f.x;
            A[kk][2 * j + 1] = (__bf16)f.y;
        }
    }

    float sacc[8];
#pragma unroll
    for (int r = 0; r < 8; ++r) sacc[r] = 0.f;

    // 16 iterations, 32 attention columns (two 16-wide a-tiles) each
    for (int apair = 0; apair < ATTz / 32; ++apair) {
        const int a0 = apair * 32;
        __syncthreads();
        // Cooperative staging: 32 rows x 512 cols of W_enc -> bf16, pre-swizzled
        // into B-fragment order. Aligned float4 groups never cross the swizzle
        // boundaries, so each float4 becomes one 8-byte LDS store.
        for (int i4 = tid; i4 < 32 * 128; i4 += 256) {
            int r  = i4 >> 7;              // row within the 32-row dual tile
            int k  = (i4 & 127) << 2;      // column, multiple of 4
            int t  = r >> 4;               // which 16-row tile
            int rr = r & 15;
            float4 f = *(const float4*)(Wenc + (size_t)(a0 + r) * ENCz + k);
            int kk  = k >> 5;
            int k32 = k & 31;
            int ln  = rr + (((k32 >> 3) & 1) << 4);
            int idx = (k32 & 7) + ((k32 >> 4) << 3);
            v4bf p;
            p[0] = (__bf16)f.x; p[1] = (__bf16)f.y;
            p[2] = (__bf16)f.z; p[3] = (__bf16)f.w;
            *(v4bf*)&ldsB[t][(kk * 32 + ln) * 16 + idx] = p;
        }
        __syncthreads();

        v8f c0 = {};
        v8f c1 = {};
#pragma unroll
        for (int kk = 0; kk < 16; ++kk) {
            v16bf B0 = *(const v16bf*)&ldsB[0][(kk * 32 + lane) * 16];
            v16bf B1 = *(const v16bf*)&ldsB[1][(kk * 32 + lane) * 16];
            c0 = __builtin_amdgcn_wmma_f32_16x16x32_bf16(
                     false, A[kk], false, B0, (short)0, c0, false, false);
            c1 = __builtin_amdgcn_wmma_f32_16x16x32_bf16(
                     false, A[kk], false, B1, (short)0, c1, false, false);
        }

        // Epilogue: C layout -> VGPR r, lanes 0-15: (M=r, N=lane); 16-31: (M=r+8, N=lane-16)
        float w0  = Watt[a0 + m];
        float w1  = Watt[a0 + 16 + m];
        float a20 = att2[(size_t)b * ATTz + a0 + m];
        float a21 = att2[(size_t)b * ATTz + a0 + 16 + m];
#pragma unroll
        for (int r = 0; r < 8; ++r) {
            float h0 = c0[r] + a20;
            float h1 = c1[r] + a21;
            h0 = h0 > 0.f ? h0 : 0.f;
            h1 = h1 > 0.f ? h1 : 0.f;
            sacc[r] = fmaf(h0, w0, sacc[r]);
            sacc[r] = fmaf(h1, w1, sacc[r]);
        }
    }

    // Reduce over the 16 N-lanes of each half (masks 1,2,4,8 stay within halves)
    float bb = batt[0];
#pragma unroll
    for (int r = 0; r < 8; ++r) {
        float v = sacc[r];
        v += __shfl_xor(v, 1);
        v += __shfl_xor(v, 2);
        v += __shfl_xor(v, 4);
        v += __shfl_xor(v, 8);
        if (m == 0) scores[(size_t)b * Sz + s0 + r + hi * 8] = v + bb;
    }
}

// ---------------------------------------------------------------------------
// Kernel 3: softmax over S per batch. grid(B) x 256, 16 elements/thread.
// ---------------------------------------------------------------------------
__global__ __launch_bounds__(256) void softmax_kernel(const float* __restrict__ scores,
                                                      float* __restrict__ alpha) {
    const int b = blockIdx.x, tid = threadIdx.x;
    const int wave = tid >> 5, lane = tid & 31;
    const float* sc = scores + (size_t)b * Sz;
    __shared__ float smax[8], ssum[8];

    float v[16];
    float m = -INFINITY;
#pragma unroll
    for (int i = 0; i < 16; ++i) {
        v[i] = sc[tid + i * 256];
        m = fmaxf(m, v[i]);
    }
    for (int off = 1; off < 32; off <<= 1) m = fmaxf(m, __shfl_xor(m, off));
    if (lane == 0) smax[wave] = m;
    __syncthreads();
    float M = smax[0];
#pragma unroll
    for (int w = 1; w < 8; ++w) M = fmaxf(M, smax[w]);

    float s = 0.f;
#pragma unroll
    for (int i = 0; i < 16; ++i) {
        v[i] = __expf(v[i] - M);
        s += v[i];
    }
    for (int off = 1; off < 32; off <<= 1) s += __shfl_xor(s, off);
    if (lane == 0) ssum[wave] = s;
    __syncthreads();
    float S = 0.f;
#pragma unroll
    for (int w = 0; w < 8; ++w) S += ssum[w];
    float inv = 1.0f / S;
#pragma unroll
    for (int i = 0; i < 16; ++i)
        alpha[(size_t)b * Sz + tid + i * 256] = v[i] * inv;
}

// ---------------------------------------------------------------------------
// Kernel 4: context partials. grid (16, B) x 256; each block: 256 S-rows.
// ---------------------------------------------------------------------------
__global__ __launch_bounds__(256) void ctx_partial_kernel(const float* __restrict__ enc,
                                                          const float* __restrict__ alpha,
                                                          float* __restrict__ partial) {
    const int b = blockIdx.y, c = blockIdx.x, tid = threadIdx.x;
    const int s0 = c * 256;
    const float* ep = enc + (((size_t)b * Sz) + s0) * (size_t)ENCz;
    const float* ap = alpha + (size_t)b * Sz + s0;
    float a0 = 0.f, a1 = 0.f;
    for (int s = 0; s < 256; ++s) {
        const float* row = ep + (size_t)s * ENCz;
        __builtin_prefetch(row + 8 * ENCz, 0, 1);   // global_prefetch_b8
        float al = ap[s];
        a0 = fmaf(al, row[tid], a0);
        a1 = fmaf(al, row[tid + 256], a1);
    }
    float* out = partial + ((size_t)(b * 16 + c)) * ENCz;
    out[tid] = a0;
    out[tid + 256] = a1;
}

// ---------------------------------------------------------------------------
// Kernel 5: deterministic reduction of partials -> context. grid(B) x 512.
// ---------------------------------------------------------------------------
__global__ __launch_bounds__(512) void ctx_reduce_kernel(const float* __restrict__ partial,
                                                         float* __restrict__ ctx) {
    const int b = blockIdx.x, e = threadIdx.x;
    float s = 0.f;
#pragma unroll
    for (int c = 0; c < 16; ++c)
        s += partial[((size_t)(b * 16 + c)) * ENCz + e];
    ctx[(size_t)b * ENCz + e] = s;
}

// ---------------------------------------------------------------------------
extern "C" void kernel_launch(void* const* d_in, const int* in_sizes, int n_in,
                              void* d_out, int out_size, void* d_ws, size_t ws_size,
                              hipStream_t stream) {
    (void)in_sizes; (void)n_in; (void)out_size; (void)ws_size;
    const float* enc  = (const float*)d_in[0];   // [B,S,ENC]
    const float* dec  = (const float*)d_in[1];   // [B,DEC]
    const float* Wenc = (const float*)d_in[2];   // [ATT,ENC]
    const float* Whid = (const float*)d_in[3];   // [ATT,DEC]
    const float* Watt = (const float*)d_in[4];   // [1,ATT]
    const float* batt = (const float*)d_in[5];   // [1]

    float* out_ctx   = (float*)d_out;                       // [B,ENC]
    float* out_alpha = (float*)d_out + (size_t)Bz * ENCz;   // [B,S]

    // workspace layout
    float* ws_att2    = (float*)d_ws;                                // 32*512
    float* ws_scores  = ws_att2 + (size_t)Bz * ATTz;                 // 32*4096
    float* ws_partial = ws_scores + (size_t)Bz * Sz;                 // 32*16*512

    {
        dim3 g(ATTz / 256, Bz);
        att2_kernel<<<g, 256, 0, stream>>>(dec, Whid, ws_att2);
    }
    {
        dim3 g(Sz / 128, Bz);
        scores_kernel<<<g, 256, 0, stream>>>(enc, Wenc, Watt, batt, ws_att2, ws_scores);
    }
    softmax_kernel<<<Bz, 256, 0, stream>>>(ws_scores, out_alpha);
    {
        dim3 g(16, Bz);
        ctx_partial_kernel<<<g, 256, 0, stream>>>(enc, out_alpha, ws_partial);
    }
    ctx_reduce_kernel<<<Bz, 512, 0, stream>>>(ws_partial, out_ctx);
}
